// HAWPv2_10960756539488
// MI455X (gfx1250) — compile-verified, matrix operand order checked by reading, not run on hardware
//
#include <hip/hip_runtime.h>
#include <hip/hip_bf16.h>
#include <math.h>

#define HH 128
#define WW 128
#define NPIX 16384
#define KJ 300
#define NLINES 81920            // 5 * 128 * 128
#define NPAIRS 90000            // 300 * 300
#define BIGV 81921              // n + 1
#define FPAD 512                // 496 padded to mult of 32
#define CHUNK 8192
#define PI_F 3.14159265358979323846f

// GEMM tiling
#define TM 64                   // workgroup M tile
#define TN 128                  // workgroup N tile
#define TK 32                   // K step (one WMMA depth)

typedef __attribute__((ext_vector_type(16))) _Float16 v16h;
typedef __attribute__((ext_vector_type(8)))  _Float16 v8h;
typedef __attribute__((ext_vector_type(8)))  float    v8f;
typedef __attribute__((ext_vector_type(4)))  unsigned int u32x4;
typedef __attribute__((ext_vector_type(8)))  int          i32x8;
typedef __attribute__((ext_vector_type(4)))  int          i32x4;

__device__ __forceinline__ float sigmoidf_(float x) { return 1.0f / (1.0f + expf(-x)); }

// ---------------- heatmap processing ----------------
__global__ void k_jloc(const float* __restrict__ hm, float* __restrict__ jloc) {
    int p = blockIdx.x * 256 + threadIdx.x;
    if (p >= NPIX) return;
    float a = hm[5 * NPIX + p], b = hm[6 * NPIX + p];
    float m = fmaxf(a, b);
    float ea = expf(a - m), eb = expf(b - m);
    jloc[p] = eb / (ea + eb);
}

__global__ void k_nms(const float* __restrict__ jloc, float* __restrict__ nms) {
    int p = blockIdx.x * 256 + threadIdx.x;
    if (p >= NPIX) return;
    int y = p >> 7, x = p & 127;
    float v = jloc[p], mx = -INFINITY;
    for (int dy = -1; dy <= 1; ++dy) {
        int yy = y + dy; if (yy < 0 || yy >= HH) continue;
        for (int dx = -1; dx <= 1; ++dx) {
            int xx = x + dx; if (xx < 0 || xx >= WW) continue;
            mx = fmaxf(mx, jloc[yy * WW + xx]);
        }
    }
    nms[p] = (v == mx) ? v : 0.0f;
}

// single block, 256 threads, iterative top-K with first-index tiebreak
__global__ void k_topk(const float* __restrict__ nms, const float* __restrict__ hm,
                       float* __restrict__ junc) {
    __shared__ float sv[256];
    __shared__ int   si[256];
    int tid = threadIdx.x;
    float loc[64];
    #pragma unroll
    for (int s = 0; s < 64; ++s) loc[s] = nms[s * 256 + tid];
    float bv = -INFINITY; int bi = 1 << 30;
    for (int s = 0; s < 64; ++s) {
        int gi = s * 256 + tid;
        if (loc[s] > bv || (loc[s] == bv && gi < bi)) { bv = loc[s]; bi = gi; }
    }
    for (int k = 0; k < KJ; ++k) {
        sv[tid] = bv; si[tid] = bi;
        __syncthreads();
        for (int off = 128; off > 0; off >>= 1) {
            if (tid < off) {
                float ov = sv[tid + off]; int oi = si[tid + off];
                if (ov > sv[tid] || (ov == sv[tid] && oi < si[tid])) { sv[tid] = ov; si[tid] = oi; }
            }
            __syncthreads();
        }
        int gi = si[0];
        __syncthreads();
        if (tid == 0) {
            int y = gi >> 7, x = gi & 127;
            junc[2 * k]     = (float)x + sigmoidf_(hm[7 * NPIX + gi]);
            junc[2 * k + 1] = (float)y + sigmoidf_(hm[8 * NPIX + gi]);
        }
        if ((gi & 255) == tid) {
            loc[gi >> 8] = -INFINITY;
            bv = -INFINITY; bi = 1 << 30;
            for (int s = 0; s < 64; ++s) {
                int g2 = s * 256 + tid;
                if (loc[s] > bv || (loc[s] == bv && g2 < bi)) { bv = loc[s]; bi = g2; }
            }
        }
        __syncthreads();
    }
}

// ---------------- HAFM decode ----------------
__global__ void k_decode(const float* __restrict__ hm, float* __restrict__ lines) {
    int i = blockIdx.x * 256 + threadIdx.x;
    if (i >= NLINES) return;
    int b = i >> 14, p = i & (NPIX - 1);
    int y = p >> 7, x = p & 127;
    float a0 = sigmoidf_(hm[0 * NPIX + p]);
    float a1 = sigmoidf_(hm[1 * NPIX + p]);
    float a2 = sigmoidf_(hm[2 * NPIX + p]);
    float dist = sigmoidf_(hm[3 * NPIX + p]);
    float bias = sigmoidf_(hm[4 * NPIX + p]);
    float d = fminf(fmaxf(dist + bias * (float)(b - 2), 0.0f), 1.0f);
    float ang0 = (a0 - 0.5f) * (2.0f * PI_F);
    float c0 = cosf(ang0), s0 = sinf(ang0);
    float t1 = tanf(a1 * PI_F * 0.5f);
    float t2 = tanf(-a2 * PI_F * 0.5f);
    float ds = d * 5.0f;
    float x1 = fminf(fmaxf((c0 - s0 * t1) * ds + (float)x, 0.0f), 127.0f);
    float y1 = fminf(fmaxf((s0 + c0 * t1) * ds + (float)y, 0.0f), 127.0f);
    float x2 = fminf(fmaxf((c0 - s0 * t2) * ds + (float)x, 0.0f), 127.0f);
    float y2 = fminf(fmaxf((s0 + c0 * t2) * ds + (float)y, 0.0f), 127.0f);
    lines[i * 4 + 0] = x1; lines[i * 4 + 1] = y1;
    lines[i * 4 + 2] = x2; lines[i * 4 + 3] = y2;
}

// ---------------- pairing ----------------
__global__ void k_fill_i32(int* __restrict__ d, int v, int n) {
    int i = blockIdx.x * 256 + threadIdx.x;
    if (i < n) d[i] = v;
}

__global__ void k_propose(const float* __restrict__ lines, const float* __restrict__ junc,
                          int* __restrict__ first) {
    int i = blockIdx.x * 256 + threadIdx.x;
    if (i >= NLINES) return;
    float x1 = lines[i * 4], y1 = lines[i * 4 + 1];
    float x2 = lines[i * 4 + 2], y2 = lines[i * 4 + 3];
    float d1 = INFINITY, d2 = INFINITY; int j1 = 0, j2 = 0;
    for (int j = 0; j < KJ; ++j) {
        float jx = junc[2 * j], jy = junc[2 * j + 1];
        float dx = jx - x1, dy = jy - y1;
        float d = dx * dx + dy * dy;
        if (d < d1) { d1 = d; j1 = j; }
        dx = jx - x2; dy = jy - y2;
        d = dx * dx + dy * dy;
        if (d < d2) { d2 = d; j2 = j; }
    }
    int a = min(j1, j2), b = max(j1, j2);
    bool keep = (a < b) && (d1 < 10.0f) && (d2 < 10.0f);
    int pid = keep ? (a * KJ + b) : (KJ * KJ);
    atomicMin(&first[pid], i);
}

__global__ void k_pairs(const int* __restrict__ first, const float* __restrict__ lines,
                        const float* __restrict__ junc, float* __restrict__ out_coords,
                        float* __restrict__ out_valid, float* __restrict__ lines_raw) {
    int p = blockIdx.x * 256 + threadIdx.x;
    if (p >= NPAIRS) return;
    int f = first[p];
    out_valid[p] = (f < BIGV) ? 1.0f : 0.0f;
    int idx = min(f, NLINES - 1);
    lines_raw[p * 4 + 0] = lines[idx * 4 + 0];
    lines_raw[p * 4 + 1] = lines[idx * 4 + 1];
    lines_raw[p * 4 + 2] = lines[idx * 4 + 2];
    lines_raw[p * 4 + 3] = lines[idx * 4 + 3];
    int pa = p / KJ, pb = p % KJ;
    out_coords[p * 4 + 0] = junc[2 * pa];
    out_coords[p * 4 + 1] = junc[2 * pa + 1];
    out_coords[p * 4 + 2] = junc[2 * pb];
    out_coords[p * 4 + 3] = junc[2 * pb + 1];
}

// ---------------- conversions / packing ----------------
// features (256 x NPIX) f32  ->  featT (NPIX x 256) f16
__global__ void k_feat_t(const float* __restrict__ src, _Float16* __restrict__ dst) {
    int i = blockIdx.x * 256 + threadIdx.x;
    if (i >= NPIX * 256) return;
    int p = i >> 8, c = i & 255;
    dst[i] = (_Float16)src[c * NPIX + p];
}

__global__ void k_pack(const float* __restrict__ src, _Float16* __restrict__ dst,
                       int rows, int ks, int kd) {
    int i = blockIdx.x * 256 + threadIdx.x;
    int n = rows * kd;
    if (i >= n) return;
    int r = i / kd, k = i - r * kd;
    dst[i] = (k < ks) ? (_Float16)src[r * ks + k] : (_Float16)0.0f;
}

__global__ void k_zero_h(_Float16* __restrict__ d, int n) {
    int i = blockIdx.x * 256 + threadIdx.x;
    if (i < n) d[i] = (_Float16)0.0f;
}

__global__ void k_projbias(const float* __restrict__ b1, const float* __restrict__ b3,
                           const float* __restrict__ b4, float* __restrict__ out) {
    int i = threadIdx.x;
    if (i >= 144) return;
    float v = 0.0f;
    if (i < 128) v = b1[i];
    else if (i < 132) v = b3[i - 128];
    else if (i < 136) v = b4[i - 132];
    out[i] = v;
}

// ---------------- TDM: 2D tile load into LDS ----------------
// Loads a (tile_rows x TK) f16 tile starting at gptr (row stride = ld elements)
// into LDS at byte offset lds_off. rem_rows rows are readable from tile start;
// rows beyond rem_rows are zero-filled by TDM OOB semantics.
__device__ __forceinline__ void tdm_load_2d(unsigned lds_off, const _Float16* gptr,
                                            int tile_rows, int rem_rows, int ld) {
    unsigned long long ga = (unsigned long long)gptr;
    if (rem_rows < 0) rem_rows = 0;
    u32x4 g0;
    g0[0] = 1u;                                            // count=1, user descriptor
    g0[1] = lds_off;                                       // lds_addr (bytes)
    g0[2] = (unsigned)(ga & 0xFFFFFFFFu);                  // global_addr[31:0]
    g0[3] = (unsigned)((ga >> 32) & 0x01FFFFFFu)           // global_addr[56:32]
          | (2u << 30);                                    // type = 2 (image)
    i32x8 g1;
    g1[0] = (1 << 16);                                     // wg_mask=0, data_size=1 (2B)
    g1[1] = (int)((unsigned)(ld & 0xFFFF) << 16);          // tensor_dim0[15:0]
    g1[2] = (int)(((unsigned)ld >> 16) & 0xFFFFu)          // tensor_dim0[31:16]
          | (int)((unsigned)(rem_rows & 0xFFFF) << 16);    // tensor_dim1[15:0]
    g1[3] = (int)(((unsigned)rem_rows >> 16) & 0xFFFFu)    // tensor_dim1[31:16]
          | (TK << 16);                                    // tile_dim0 = 32
    g1[4] = (tile_rows & 0xFFFF);                          // tile_dim1, tile_dim2=0
    g1[5] = ld;                                            // tensor_dim0_stride[31:0]
    g1[6] = 0;                                             // stride hi, dim1_stride lo
    g1[7] = 0;
    i32x4 z4 = {0, 0, 0, 0};
#if defined(__clang_major__) && (__clang_major__ >= 23)
    i32x8 z8 = {0, 0, 0, 0, 0, 0, 0, 0};
    __builtin_amdgcn_tensor_load_to_lds(g0, g1, z4, z4, z8, 0);
#else
    __builtin_amdgcn_tensor_load_to_lds(g0, g1, z4, z4, 0);
#endif
}

// fragment from an LDS tile stored row-major with 32-half rows
__device__ __forceinline__ v16h frag_lds(const _Float16* tile, int row0, int l15, int kk) {
    const _Float16* p = tile + (row0 + l15) * TK + kk;
    v8h lo = *(const v8h*)p;
    v8h hi = *(const v8h*)(p + 16);
    v16h f;
    #pragma unroll
    for (int i = 0; i < 8; ++i) { f[i] = lo[i]; f[8 + i] = hi[i]; }
    return f;
}

// ---------------- WMMA GEMM with TDM double-buffered LDS staging ----------------
// C[M,N] = A[M,K] * W^T + bias.  A row-major f16 (lda), B = W row-major N x K (ldb).
// Workgroup tile 64(M) x 128(N); 8 waves, each computes 32x32 via 4 WMMAs/K-step.
template<int BIAS_MODE, int RELU, int OUT_F16>
__global__ __launch_bounds__(256) void k_gemm(const _Float16* __restrict__ A,
                                              const _Float16* __restrict__ B,
                                              const float* __restrict__ bias,
                                              void* __restrict__ Cp,
                                              int M, int N, int K,
                                              int lda, int ldb, int ldc) {
    __shared__ _Float16 sA[2][TM * TK];      // 2 x 4KB
    __shared__ _Float16 sB[2][TN * TK];      // 2 x 8KB
    int tid = threadIdx.x;
    int lane = tid & 31, wave = tid >> 5;
    int wm = wave >> 2;                      // 0..1 : 32-row sub-tile
    int wn = wave & 3;                       // 0..3 : 32-col sub-tile
    int half = lane >> 4;
    int l15 = lane & 15;
    int kk = half * 8;

    int rowBase = blockIdx.y * TM;
    int colBase = blockIdx.x * TN;

    unsigned offA[2] = { (unsigned)(unsigned long long)(const void*)&sA[0][0],
                         (unsigned)(unsigned long long)(const void*)&sA[1][0] };
    unsigned offB[2] = { (unsigned)(unsigned long long)(const void*)&sB[0][0],
                         (unsigned)(unsigned long long)(const void*)&sB[1][0] };

    int nsteps = K / TK;
    // prologue: stage first K-slice
    if (wave == 0) tdm_load_2d(offA[0], A + (long)rowBase * lda, TM, M - rowBase, lda);
    if (wave == 1) tdm_load_2d(offB[0], B + (long)colBase * ldb, TN, N - colBase, ldb);
    if (wave <= 1) __builtin_amdgcn_s_wait_tensorcnt(0);
    __syncthreads();

    v8f acc00 = {}, acc01 = {}, acc10 = {}, acc11 = {};
    for (int s = 0; s < nsteps; ++s) {
        int cur = s & 1, nxt = cur ^ 1;
        if (s + 1 < nsteps) {
            int k0 = (s + 1) * TK;
            if (wave == 0) tdm_load_2d(offA[nxt], A + (long)rowBase * lda + k0, TM, M - rowBase, lda);
            if (wave == 1) tdm_load_2d(offB[nxt], B + (long)colBase * ldb + k0, TN, N - colBase, ldb);
        }
        v16h a0 = frag_lds(&sA[cur][0], wm * 32,      l15, kk);
        v16h a1 = frag_lds(&sA[cur][0], wm * 32 + 16, l15, kk);
        v16h b0 = frag_lds(&sB[cur][0], wn * 32,      l15, kk);
        v16h b1 = frag_lds(&sB[cur][0], wn * 32 + 16, l15, kk);
        acc00 = __builtin_amdgcn_wmma_f32_16x16x32_f16(false, a0, false, b0, (short)0, acc00, false, false);
        acc01 = __builtin_amdgcn_wmma_f32_16x16x32_f16(false, a0, false, b1, (short)0, acc01, false, false);
        acc10 = __builtin_amdgcn_wmma_f32_16x16x32_f16(false, a1, false, b0, (short)0, acc10, false, false);
        acc11 = __builtin_amdgcn_wmma_f32_16x16x32_f16(false, a1, false, b1, (short)0, acc11, false, false);
        if (s + 1 < nsteps) {
            if (wave <= 1) __builtin_amdgcn_s_wait_tensorcnt(0);
            __syncthreads();
        }
    }

    // epilogue: whole-tile fast path, guarded tail
    bool fullM = (rowBase + TM <= M);
    #pragma unroll
    for (int mi = 0; mi < 2; ++mi) {
        #pragma unroll
        for (int ni = 0; ni < 2; ++ni) {
            v8f acc = (mi == 0) ? (ni == 0 ? acc00 : acc01) : (ni == 0 ? acc10 : acc11);
            int rbase = rowBase + wm * 32 + mi * 16 + half * 8;
            int col = colBase + wn * 32 + ni * 16 + l15;
            float bn = (BIAS_MODE == 1) ? bias[col] : 0.0f;
            if (fullM) {
                #pragma unroll
                for (int i = 0; i < 8; ++i) {
                    int row = rbase + i;
                    float v = acc[i] + bn;
                    if (BIAS_MODE == 2) v += bias[row];
                    if (RELU) v = fmaxf(v, 0.0f);
                    if (OUT_F16) ((_Float16*)Cp)[(long)row * ldc + col] = (_Float16)v;
                    else         ((float*)Cp)[(long)row * ldc + col] = v;
                }
            } else {
                #pragma unroll
                for (int i = 0; i < 8; ++i) {
                    int row = rbase + i;
                    if (row < M) {
                        float v = acc[i] + bn;
                        if (BIAS_MODE == 2) v += bias[row];
                        if (RELU) v = fmaxf(v, 0.0f);
                        if (OUT_F16) ((_Float16*)Cp)[(long)row * ldc + col] = (_Float16)v;
                        else         ((float*)Cp)[(long)row * ldc + col] = v;
                    }
                }
            }
        }
    }
}

// ---------------- bilinear sampling ----------------
__device__ __forceinline__ float bilin(const float* __restrict__ f, float x, float y) {
    x = fminf(fmaxf(x, 0.0f), 127.0f);
    y = fminf(fmaxf(y, 0.0f), 127.0f);
    float x0f = floorf(x), y0f = floorf(y);
    int x0 = (int)x0f, y0 = (int)y0f;
    int x1 = min(x0 + 1, 127), y1 = min(y0 + 1, 127);
    float wx = x - x0f, wy = y - y0f;
    float f00 = f[y0 * WW + x0], f01 = f[y0 * WW + x1];
    float f10 = f[y1 * WW + x0], f11 = f[y1 * WW + x1];
    return f00 * (1.0f - wx) * (1.0f - wy) + f01 * wx * (1.0f - wy)
         + f10 * (1.0f - wx) * wy + f11 * wx * wy;
}

__global__ void k_endpoints(const float* __restrict__ coords, const float* __restrict__ proj,
                            _Float16* __restrict__ lf) {
    int p = blockIdx.x, c = threadIdx.x;     // 128 threads = channels
    float x1 = coords[p * 4 + 0] - 0.5f, y1 = coords[p * 4 + 1] - 0.5f;
    float x2 = coords[p * 4 + 2] - 0.5f, y2 = coords[p * 4 + 3] - 0.5f;
    const float* f = proj + (long)c * NPIX;
    _Float16* row = lf + (long)p * FPAD;
    row[c]       = (_Float16)bilin(f, x1, y1);
    row[128 + c] = (_Float16)bilin(f, x2, y2);
    if (c < 16) row[496 + c] = (_Float16)0.0f;   // K pad
}

__global__ void k_loi(const float* __restrict__ coords, const float* __restrict__ lines_raw,
                      const float* __restrict__ proj, _Float16* __restrict__ lf) {
    int p = blockIdx.x, tid = threadIdx.x;
    if (tid >= 240) return;
    int isaux = tid >= 120;
    int t = isaux ? tid - 120 : tid;
    int c = t / 30, si = t % 30;
    const float* src = isaux ? (lines_raw + p * 4) : (coords + p * 4);
    float tt = (float)(si + 1) / 31.0f;
    float x = src[0] * (1.0f - tt) + src[2] * tt;
    float y = src[1] * (1.0f - tt) + src[3] * tt;
    int ch = (isaux ? 132 : 128) + c;
    float v = bilin(proj + (long)ch * NPIX, x, y);
    lf[(long)p * FPAD + (isaux ? 376 : 256) + c * 30 + si] = (_Float16)v;
}

// ---------------- final head: softmax over 2 logits ----------------
__global__ void k_logits(const _Float16* __restrict__ h3, const _Float16* __restrict__ r,
                         const float* __restrict__ Wh, const float* __restrict__ bh,
                         const float* __restrict__ validf, float* __restrict__ scores,
                         int rows, int base) {
    int row = blockIdx.x * 8 + (threadIdx.x >> 5);
    int lane = threadIdx.x & 31;
    if (row >= rows) return;
    const _Float16* hp = h3 + (long)row * 1024;
    const _Float16* rp = r  + (long)row * 1024;
    float s0 = 0.0f, s1 = 0.0f;
    for (int k = lane; k < 1024; k += 32) {
        float v = (float)hp[k] + (float)rp[k];
        s0 += v * Wh[k];
        s1 += v * Wh[1024 + k];
    }
    for (int off = 16; off > 0; off >>= 1) {
        s0 += __shfl_down(s0, off, 32);
        s1 += __shfl_down(s1, off, 32);
    }
    if (lane == 0) {
        float l0 = s0 + bh[0], l1 = s1 + bh[1];
        float m = fmaxf(l0, l1);
        float e0 = expf(l0 - m), e1 = expf(l1 - m);
        int g = base + row;
        scores[g] = (e1 / (e0 + e1)) * validf[g];
    }
}

extern "C" void kernel_launch(void* const* d_in, const int* in_sizes, int n_in,
                              void* d_out, int out_size, void* d_ws, size_t ws_size,
                              hipStream_t stream) {
    (void)in_sizes; (void)n_in; (void)out_size; (void)ws_size;
    const float* features = (const float*)d_in[0];
    const float* hm    = (const float*)d_in[1];
    const float* fc1_w = (const float*)d_in[2];  const float* fc1_b = (const float*)d_in[3];
    const float* fc3_w = (const float*)d_in[4];  const float* fc3_b = (const float*)d_in[5];
    const float* fc4_w = (const float*)d_in[6];  const float* fc4_b = (const float*)d_in[7];
    const float* W0 = (const float*)d_in[8];     const float* b0 = (const float*)d_in[9];
    const float* W1 = (const float*)d_in[10];    const float* b1 = (const float*)d_in[11];
    const float* W2 = (const float*)d_in[12];    const float* b2 = (const float*)d_in[13];
    const float* Wr = (const float*)d_in[14];    const float* br = (const float*)d_in[15];
    const float* Wh = (const float*)d_in[16];    const float* bh = (const float*)d_in[17];

    float* out_coords = (float*)d_out;               // 90000 x 4
    float* out_scores = out_coords + 4 * NPAIRS;     // 90000
    float* out_valid  = out_scores + NPAIRS;         // 90000

    char* w = (char*)d_ws;
    auto alloc = [&](size_t b) -> char* {
        char* p = w; w += (b + 255) & ~(size_t)255; return p;
    };
    float*     jloc      = (float*)alloc((size_t)NPIX * 4);
    float*     nms       = (float*)alloc((size_t)NPIX * 4);
    float*     junc      = (float*)alloc((size_t)KJ * 2 * 4);
    float*     lines_all = (float*)alloc((size_t)NLINES * 4 * 4);
    int*       first     = (int*)alloc((size_t)(NPAIRS + 1) * 4);
    float*     lines_raw = (float*)alloc((size_t)NPAIRS * 4 * 4);
    _Float16*  featT     = (_Float16*)alloc((size_t)NPIX * 256 * 2);   // NPIX x 256
    _Float16*  projW     = (_Float16*)alloc((size_t)192 * 256 * 2);    // padded to tile
    float*     projB     = (float*)alloc(192 * 4);
    float*     proj_out  = (float*)alloc((size_t)136 * NPIX * 4);
    _Float16*  W0h       = (_Float16*)alloc((size_t)1024 * 512 * 2);
    _Float16*  W1h       = (_Float16*)alloc((size_t)1024 * 1024 * 2);
    _Float16*  W2h       = (_Float16*)alloc((size_t)1024 * 1024 * 2);
    _Float16*  Wrh       = (_Float16*)alloc((size_t)1024 * 256 * 2);
    _Float16*  lf        = (_Float16*)alloc((size_t)NPAIRS * FPAD * 2);
    _Float16*  hA        = (_Float16*)alloc((size_t)CHUNK * 1024 * 2);
    _Float16*  hB        = (_Float16*)alloc((size_t)CHUNK * 1024 * 2);

    dim3 B256(256);
    // ---- stage 1: junctions + lines ----
    k_jloc<<<dim3((NPIX + 255) / 256), B256, 0, stream>>>(hm, jloc);
    k_nms<<<dim3((NPIX + 255) / 256), B256, 0, stream>>>(jloc, nms);
    k_topk<<<dim3(1), B256, 0, stream>>>(nms, hm, junc);
    k_decode<<<dim3((NLINES + 255) / 256), B256, 0, stream>>>(hm, lines_all);
    k_fill_i32<<<dim3((NPAIRS + 256) / 256), B256, 0, stream>>>(first, BIGV, NPAIRS + 1);
    k_propose<<<dim3((NLINES + 255) / 256), B256, 0, stream>>>(lines_all, junc, first);
    k_pairs<<<dim3((NPAIRS + 255) / 256), B256, 0, stream>>>(first, lines_all, junc,
                                                             out_coords, out_valid, lines_raw);

    // ---- stage 2: f16 conversions / weight packing ----
    k_feat_t<<<dim3((NPIX * 256 + 255) / 256), B256, 0, stream>>>(features, featT);
    k_pack<<<dim3((128 * 256 + 255) / 256), B256, 0, stream>>>(fc1_w, projW, 128, 256, 256);
    k_pack<<<dim3((4 * 256 + 255) / 256), B256, 0, stream>>>(fc3_w, projW + 128 * 256, 4, 256, 256);
    k_pack<<<dim3((4 * 256 + 255) / 256), B256, 0, stream>>>(fc4_w, projW + 132 * 256, 4, 256, 256);
    k_zero_h<<<dim3((56 * 256 + 255) / 256), B256, 0, stream>>>(projW + 136 * 256, 56 * 256);
    k_projbias<<<dim3(1), dim3(192), 0, stream>>>(fc1_b, fc3_b, fc4_b, projB);
    k_pack<<<dim3((1024 * 512 + 255) / 256), B256, 0, stream>>>(W0, W0h, 1024, 496, 512);
    k_pack<<<dim3((1024 * 1024 + 255) / 256), B256, 0, stream>>>(W1, W1h, 1024, 1024, 1024);
    k_pack<<<dim3((1024 * 1024 + 255) / 256), B256, 0, stream>>>(W2, W2h, 1024, 1024, 1024);
    k_pack<<<dim3((1024 * 256 + 255) / 256), B256, 0, stream>>>(Wr, Wrh, 1024, 240, 256);

    // ---- stage 3: feature projections (loi/thinf/auxf) via WMMA ----
    // M=136 channels (A=projW), N=16384 pixels (B=featT, N x K), K=256
    k_gemm<2, 0, 0><<<dim3(NPIX / TN, (136 + TM - 1) / TM), B256, 0, stream>>>(
        projW, featT, projB, proj_out, 136, NPIX, 256, 256, 256, NPIX);

    // ---- stage 4: build line features (496 -> padded 512) ----
    k_endpoints<<<dim3(NPAIRS), dim3(128), 0, stream>>>(out_coords, proj_out, lf);
    k_loi<<<dim3(NPAIRS), B256, 0, stream>>>(out_coords, lines_raw, proj_out, lf);

    // ---- stage 5: MLP via WMMA, chunked over M ----
    for (int base = 0; base < NPAIRS; base += CHUNK) {
        int rows = min(CHUNK, NPAIRS - base);
        dim3 gm(1024 / TN, (rows + TM - 1) / TM);
        // h = relu(lf @ W0.T + b0)
        k_gemm<1, 1, 1><<<gm, B256, 0, stream>>>(lf + (size_t)base * FPAD, W0h, b0,
                                                 hA, rows, 1024, 512, FPAD, 512, 1024);
        // h = relu(h @ W1.T + b1)
        k_gemm<1, 1, 1><<<gm, B256, 0, stream>>>(hA, W1h, b1,
                                                 hB, rows, 1024, 1024, 1024, 1024, 1024);
        // h = h @ W2.T + b2  (no relu)
        k_gemm<1, 0, 1><<<gm, B256, 0, stream>>>(hB, W2h, b2,
                                                 hA, rows, 1024, 1024, 1024, 1024, 1024);
        // r = relu(lf[:,256:496] @ Wr.T + br)
        k_gemm<1, 1, 1><<<gm, B256, 0, stream>>>(lf + (size_t)base * FPAD + 256, Wrh, br,
                                                 hB, rows, 1024, 256, FPAD, 256, 1024);
        // logits -> softmax[:,1] * valid
        k_logits<<<dim3((rows + 7) / 8), B256, 0, stream>>>(hA, hB, Wh, bh,
                                                            out_valid, out_scores, rows, base);
    }
}
